// BGNetMILHead_70600672412159
// MI455X (gfx1250) — matrix operand order repaired and checked by reading, not compiled, not run on hardware
//
#include <hip/hip_runtime.h>
#include <hip/hip_bf16.h>

typedef __attribute__((ext_vector_type(16))) _Float16 v16h;
typedef __attribute__((ext_vector_type(8)))  _Float16 v8h;
typedef __attribute__((ext_vector_type(8)))  float    v8f;

#define DM        512
#define NCLS      10
#define NWIN      262144
#define NBAGS     4096
#define BAG       64
#define XPAD      520      // halves per LDS row (512 + 8 pad -> 4-bank rotation)
#define LN_EPS    1e-5f
#define BTILE_H   8192     // halves per N-tile of packed B (16 kt * 32 lanes * 16)

// ---------------------------------------------------------------------------
// gfx1250 async global->LDS copy (ASYNCcnt path), with safe fallback.
// Builtin signature (from clang diagnostic): (v4i __device__*, v4i __shared__*,
// imm offset, imm cpol) where v4i = int __attribute__((vector_size(16))).
// ---------------------------------------------------------------------------
__device__ __forceinline__ void async_copy_b128(const void* g, void* l) {
#if __has_builtin(__builtin_amdgcn_global_load_async_to_lds_b128)
  typedef int v4i __attribute__((vector_size(16)));
  __builtin_amdgcn_global_load_async_to_lds_b128(
      (__attribute__((address_space(1))) v4i*)g,
      (__attribute__((address_space(3))) v4i*)l, 0, 0);
#else
  *(float4*)l = *(const float4*)g;
#endif
}

__device__ __forceinline__ void async_wait0() {
#if __has_builtin(__builtin_amdgcn_s_wait_asynccnt)
  __builtin_amdgcn_s_wait_asynccnt(0);
#endif
}

// Stage one 16KB B N-tile (all 16 kt fragments) into LDS, 128 threads.
__device__ __forceinline__ void stage_b_tile(const _Float16* __restrict__ src,
                                             _Float16* __restrict__ dst, int t) {
#pragma unroll
  for (int i = 0; i < 8; ++i) {
    const int idx = t + i * 128;               // 1024 x 16B chunks
    async_copy_b128((const char*)src + idx * 16, (char*)dst + idx * 16);
  }
}

// ---------------------------------------------------------------------------
// WMMA fragment helpers (16x16x32 f16, f32 accumulate)
// A layout (16-bit A 16x32): lanes 0-15 -> M=lane, halves = K{0..7,16..23};
//                            lanes 16-31 -> M=lane-16, halves = K{8..15,24..31}
// ---------------------------------------------------------------------------
__device__ __forceinline__ v16h load_a_frag(const _Float16* __restrict__ xh,
                                            int ldsRowBase, int kt, int lane) {
  const int m    = lane & 15;
  const int koff = kt * 32 + ((lane < 16) ? 0 : 8);
  const _Float16* p = xh + (ldsRowBase + m) * XPAD + koff;
  v8h lo = *(const v8h*)(p);
  v8h hi = *(const v8h*)(p + 16);
  v16h a;
#pragma unroll
  for (int i = 0; i < 8; ++i) { a[i] = lo[i]; a[i + 8] = hi[i]; }
  return a;
}

__device__ __forceinline__ v16h load_b_frag(const _Float16* __restrict__ bb,
                                            int kt, int lane) {
  return *(const v16h*)(bb + ((size_t)(kt * 32 + lane)) * 16);
}

__device__ __forceinline__ v8f wmma_f16(v16h a, v16h b, v8f c) {
  return __builtin_amdgcn_wmma_f32_16x16x32_f16(false, a, false, b,
                                                (short)0, c, false, false);
}

// ---------------------------------------------------------------------------
// LayerNorm of one 512-wide row, cooperative across a wave32.
// Lane holds 16 consecutive floats at column lane*16; writes f16 to LDS row.
// ---------------------------------------------------------------------------
__device__ __forceinline__ void ln_row_to_lds(const float* __restrict__ src,
                                              const float* __restrict__ g,
                                              const float* __restrict__ b,
                                              _Float16* __restrict__ dstRow,
                                              int lane) {
  const float4* p4 = (const float4*)(src + lane * 16);
  float4 v[4];
#pragma unroll
  for (int i = 0; i < 4; ++i) v[i] = p4[i];

  float s = 0.f, ss = 0.f;
#pragma unroll
  for (int i = 0; i < 4; ++i) {
    s  += v[i].x + v[i].y + v[i].z + v[i].w;
    ss += v[i].x * v[i].x + v[i].y * v[i].y + v[i].z * v[i].z + v[i].w * v[i].w;
  }
#pragma unroll
  for (int d = 16; d >= 1; d >>= 1) {
    s  += __shfl_xor(s,  d, 32);
    ss += __shfl_xor(ss, d, 32);
  }
  const float mean = s * (1.0f / DM);
  const float var  = ss * (1.0f / DM) - mean * mean;
  const float rstd = rsqrtf(var + LN_EPS);

  const float4* g4 = (const float4*)(g + lane * 16);
  const float4* b4 = (const float4*)(b + lane * 16);
  v8h o0, o1;
#pragma unroll
  for (int i = 0; i < 4; ++i) {
    float4 gg = g4[i], bb = b4[i];
    float r0 = (v[i].x - mean) * rstd * gg.x + bb.x;
    float r1 = (v[i].y - mean) * rstd * gg.y + bb.y;
    float r2 = (v[i].z - mean) * rstd * gg.z + bb.z;
    float r3 = (v[i].w - mean) * rstd * gg.w + bb.w;
    if (i < 2) {
      o0[i * 4 + 0] = (_Float16)r0; o0[i * 4 + 1] = (_Float16)r1;
      o0[i * 4 + 2] = (_Float16)r2; o0[i * 4 + 3] = (_Float16)r3;
    } else {
      o1[(i - 2) * 4 + 0] = (_Float16)r0; o1[(i - 2) * 4 + 1] = (_Float16)r1;
      o1[(i - 2) * 4 + 2] = (_Float16)r2; o1[(i - 2) * 4 + 3] = (_Float16)r3;
    }
  }
  *(v8h*)(dstRow + lane * 16)     = o0;
  *(v8h*)(dstRow + lane * 16 + 8) = o1;
}

// ---------------------------------------------------------------------------
// Kernel 0: pack W (f32 [512,512], row-major KxN) into B-fragment-native f16.
// B layout (16-bit B 32x16): lanes 0-15 -> N=lane, halves = K{0..7,16..23};
//                            lanes 16-31 -> N=lane-16, halves = K{8..15,24..31}.
// Blob index: ((nt*16 + kt)*32 + lane)*16 + h  -> per-lane contiguous 32B.
// ---------------------------------------------------------------------------
__global__ __launch_bounds__(256) void k_pack(const float* __restrict__ W1,
                                              const float* __restrict__ Wc1,
                                              _Float16* __restrict__ W1B,
                                              _Float16* __restrict__ Wc1B) {
  int tid = blockIdx.x * 256 + threadIdx.x;       // 32768 total
  const float* src; _Float16* dst; int f = tid;
  if (tid < 16384) { src = W1;  dst = W1B; }
  else             { src = Wc1; dst = Wc1B; f = tid - 16384; }
  const int lane = f & 31;
  const int kt   = (f >> 5) & 15;
  const int nt   = f >> 9;
  const int n     = nt * 16 + (lane & 15);
  const int kbase = kt * 32 + ((lane < 16) ? 0 : 8);
#pragma unroll
  for (int h = 0; h < 16; ++h)
    dst[(size_t)f * 16 + h] = (_Float16)src[(size_t)(kbase + h + ((h < 8) ? 0 : 8)) * DM + n];
}

// ---------------------------------------------------------------------------
// Kernel 1 (fused): one block == one bag of 64 contiguous windows.
//   LN (global->f16 LDS); WMMA GEMM with double-buffered async-staged B tiles;
//   tanh + dot(w2); softmax over 64 scores; pooling from L2-hot global rows.
// Dynamic LDS: xh [64*XPAD]h (66560B) + B tiles 2*16KB (32768B) + sc/wgt (512B)
//            = 99840 B -> 3 blocks/WGP, 12 waves.
// 128 threads = 4 waves; wave w owns rows w*16 .. w*16+15 (one M-tile).
// ---------------------------------------------------------------------------
__global__ __launch_bounds__(128) void k_bag(const float* __restrict__ emb,
                                             const float* __restrict__ ln_g,
                                             const float* __restrict__ ln_b,
                                             const _Float16* __restrict__ WB,
                                             const float* __restrict__ b1,
                                             const float* __restrict__ w2,
                                             const float* __restrict__ b2p,
                                             float* __restrict__ bagout) {
  extern __shared__ char smem[];
  _Float16* xh  = (_Float16*)smem;                                  // 64*XPAD
  _Float16* bt0 = (_Float16*)(smem + BAG * XPAD * 2);               // 8192 h
  _Float16* bt1 = bt0 + BTILE_H;                                    // 8192 h
  float*    sc  = (float*)(smem + BAG * XPAD * 2 + 2 * BTILE_H * 2);// [64]
  float*    wgt = sc + BAG;                                         // [64]

  const int t    = threadIdx.x;
  const int lane = t & 31;
  const int w    = t >> 5;
  const int bag  = blockIdx.x;
  const float* bagrows = emb + (size_t)bag * BAG * DM;

  // kick off async stage of B tile nt=0 (overlaps the LN phase)
  stage_b_tile(WB, bt0, t);

  // --- layernorm 16 rows per wave (from global), write f16 to xh ---
#pragma unroll 1
  for (int r = 0; r < 16; ++r) {
    const int row = w * 16 + r;
    ln_row_to_lds(bagrows + (size_t)row * DM, ln_g, ln_b, xh + row * XPAD, lane);
  }
  async_wait0();
  __syncthreads();

  // --- GEMM (16x512 tile per wave) + tanh + dot(w2) ---
  const float b2v = b2p[0];
  float sacc[8];
#pragma unroll
  for (int i = 0; i < 8; ++i) sacc[i] = 0.f;

#pragma unroll 1
  for (int nt = 0; nt < 32; ++nt) {
    const _Float16* bc = (nt & 1) ? bt1 : bt0;
    if (nt + 1 < 32)                       // prefetch next tile while computing
      stage_b_tile(WB + (size_t)(nt + 1) * BTILE_H, (nt & 1) ? bt0 : bt1, t);

    const float b1n = b1[nt * 16 + (lane & 15)];
    const float w2n = w2[nt * 16 + (lane & 15)];
    v8f c;
#pragma unroll
    for (int i = 0; i < 8; ++i) c[i] = b1n;
#pragma unroll
    for (int kt = 0; kt < 16; ++kt) {
      v16h a  = load_a_frag(xh, w * 16, kt, lane);
      v16h bf = load_b_frag(bc, kt, lane);
      c = wmma_f16(a, bf, c);
    }
#pragma unroll
    for (int i = 0; i < 8; ++i) sacc[i] += tanhf(c[i]) * w2n;

    async_wait0();       // our prefetch issued this iteration has landed
    __syncthreads();     // everyone done reading bc; next tile visible to all
  }
  // reduce the 16 N-stripes (xor within each half-wave group of 16 lanes)
#pragma unroll
  for (int d = 1; d <= 8; d <<= 1)
#pragma unroll
    for (int i = 0; i < 8; ++i) sacc[i] += __shfl_xor(sacc[i], d, 32);

  // C frag: VGPR i -> M=i (lanes 0-15) / M=8+i (lanes 16-31)
  if (lane == 0) {
#pragma unroll
    for (int i = 0; i < 8; ++i) sc[w * 16 + i] = sacc[i] + b2v;
  }
  if (lane == 16) {
#pragma unroll
    for (int i = 0; i < 8; ++i) sc[w * 16 + 8 + i] = sacc[i] + b2v;
  }
  __syncthreads();

  // --- segment softmax over the 64 scores ---
  float m = -3.4e38f;
#pragma unroll 1
  for (int i = 0; i < BAG; ++i) m = fmaxf(m, sc[i]);
  if (t < BAG) wgt[t] = expf(sc[t] - m);
  __syncthreads();
  float den = 0.f;
#pragma unroll 1
  for (int i = 0; i < BAG; ++i) den += wgt[i];
  const float inv = 1.0f / den;

  // --- weighted pooling: thread t owns dims 4t..4t+3 (rows are L2-hot) ---
  float4 acc = {0.f, 0.f, 0.f, 0.f};
#pragma unroll 1
  for (int i = 0; i < BAG; ++i) {
    const float wv = wgt[i] * inv;
    const float4 v = ((const float4*)(bagrows + (size_t)i * DM))[t];
    acc.x += wv * v.x; acc.y += wv * v.y; acc.z += wv * v.z; acc.w += wv * v.w;
  }
  ((float4*)(bagout + (size_t)bag * DM))[t] = acc;
}

// ---------------------------------------------------------------------------
// Kernel 2: classifier head. 64 blocks x 128 threads; wave owns 16 rows.
// LN2 -> WMMA GEMM (Wc1,bc1) -> exact GELU -> y (f16 LDS) -> @Wc2 + bc2.
// Dynamic LDS: xh [64*XPAD] + yh [64*XPAD] f16 = 133120 B.
// ---------------------------------------------------------------------------
__global__ __launch_bounds__(128) void k_head(const float* __restrict__ bagin,
                                              const float* __restrict__ ln_g,
                                              const float* __restrict__ ln_b,
                                              const _Float16* __restrict__ WB,
                                              const float* __restrict__ bc1,
                                              const float* __restrict__ Wc2,
                                              const float* __restrict__ bc2,
                                              float* __restrict__ logits) {
  extern __shared__ char smem[];
  _Float16* xh = (_Float16*)smem;                      // [64*XPAD]
  _Float16* yh = (_Float16*)(smem + 64 * XPAD * 2);    // [64*XPAD]

  const int t    = threadIdx.x;
  const int lane = t & 31;
  const int w    = t >> 5;
  const int rowBase = blockIdx.x * 64 + w * 16;        // global row of wave

  // --- LN2 on 16 rows per wave ---
#pragma unroll 1
  for (int r = 0; r < 16; ++r) {
    const int lrow = w * 16 + r;
    ln_row_to_lds(bagin + (size_t)(rowBase + r) * DM, ln_g, ln_b,
                  xh + lrow * XPAD, lane);
  }
  __syncthreads();

  // --- GEMM + exact GELU -> yh ---
  const float inv_sqrt2 = 0.70710678118654752f;
#pragma unroll 1
  for (int nt = 0; nt < 32; ++nt) {
    const float bn = bc1[nt * 16 + (lane & 15)];
    v8f c;
#pragma unroll
    for (int i = 0; i < 8; ++i) c[i] = bn;
#pragma unroll
    for (int kt = 0; kt < 16; ++kt) {
      v16h a  = load_a_frag(xh, w * 16, kt, lane);
      v16h bf = *(const v16h*)(WB + ((size_t)(nt * 16 + kt) * 32 + lane) * 16);
      c = wmma_f16(a, bf, c);
    }
    const int col = nt * 16 + (lane & 15);
#pragma unroll
    for (int i = 0; i < 8; ++i) {
      const int mrow = (lane < 16) ? i : (8 + i);
      float gl = 0.5f * c[i] * (1.0f + erff(c[i] * inv_sqrt2));
      yh[(w * 16 + mrow) * XPAD + col] = (_Float16)gl;
    }
  }
  __syncthreads();

  // --- logits = y @ Wc2 + bc2 (cooperative per row, shuffle reduce) ---
#pragma unroll 1
  for (int r = 0; r < 16; ++r) {
    const int lrow = w * 16 + r;
    float acc[NCLS];
#pragma unroll
    for (int c = 0; c < NCLS; ++c) acc[c] = 0.f;
#pragma unroll 1
    for (int j = 0; j < 16; ++j) {
      const int n = lane + 32 * j;
      const float yv = (float)yh[lrow * XPAD + n];
      const float* wr = Wc2 + n * NCLS;
#pragma unroll
      for (int c = 0; c < NCLS; ++c) acc[c] += yv * wr[c];
    }
#pragma unroll
    for (int d = 16; d >= 1; d >>= 1)
#pragma unroll
      for (int c = 0; c < NCLS; ++c) acc[c] += __shfl_xor(acc[c], d, 32);
    if (lane == 0) {
      float* lp = logits + (size_t)(rowBase + r) * NCLS;
#pragma unroll
      for (int c = 0; c < NCLS; ++c) lp[c] = acc[c] + bc2[c];
    }
  }
}

// ---------------------------------------------------------------------------
extern "C" void kernel_launch(void* const* d_in, const int* in_sizes, int n_in,
                              void* d_out, int out_size, void* d_ws, size_t ws_size,
                              hipStream_t stream) {
  const float* emb   = (const float*)d_in[0];
  // d_in[1] bag_slices: contiguous fixed-size bags per setup; not needed
  const float* ln1_g = (const float*)d_in[2];
  const float* ln1_b = (const float*)d_in[3];
  const float* W1    = (const float*)d_in[4];
  const float* b1    = (const float*)d_in[5];
  const float* w2    = (const float*)d_in[6];
  const float* b2    = (const float*)d_in[7];
  const float* ln2_g = (const float*)d_in[8];
  const float* ln2_b = (const float*)d_in[9];
  const float* Wc1   = (const float*)d_in[10];
  const float* bc1   = (const float*)d_in[11];
  const float* Wc2   = (const float*)d_in[12];
  const float* bc2   = (const float*)d_in[13];

  float* logits = (float*)d_out;                 // [4096,10]
  float* bagout = (float*)d_out + NBAGS * NCLS;  // [4096,512]

  char* ws = (char*)d_ws;
  _Float16* W1B  = (_Float16*)(ws);              // 512 KB
  _Float16* Wc1B = (_Float16*)(ws + 524288);     // 512 KB

  // 1) pack weights to WMMA B-fragment layout (f16)
  k_pack<<<128, 256, 0, stream>>>(W1, Wc1, W1B, Wc1B);

  // 2) fused LN + attention GEMM + tanh/w2 + softmax + pooling (one bag/block)
  const size_t lds_bag = (size_t)BAG * XPAD * 2 + (size_t)2 * BTILE_H * 2
                       + 2 * BAG * 4;
  k_bag<<<NBAGS, 128, lds_bag, stream>>>(emb, ln1_g, ln1_b, W1B, b1, w2, b2, bagout);

  // 3) classifier head
  const size_t lds_head = (size_t)2 * 64 * XPAD * 2;
  k_head<<<NBAGS / 64, 128, lds_head, stream>>>(bagout, ln2_g, ln2_b, Wc1B,
                                                bc1, Wc2, bc2, logits);
}
